// CausalSelfAttention_5162550690654
// MI455X (gfx1250) — matrix operand order repaired and checked by reading, not compiled, not run on hardware
//
#include <hip/hip_runtime.h>

// CDNA5 (gfx1250) wave32 WMMA implementation of causal self-attention.
// All three matmul stages (QKV GEMM, flash attention, output projection)
// run on v_wmma_f32_16x16x32_f16 with fp32 accumulation.
// Pure-copy tile staging uses GLOBAL_LOAD_ASYNC_TO_LDS_B128 (ASYNCcnt) when
// the toolchain exposes the builtins; otherwise falls back to b128+ds_store.

typedef __attribute__((ext_vector_type(16))) _Float16 v16h;
typedef __attribute__((ext_vector_type(8)))  float    v8f;

#define NH   16
#define NB   4
#define TT   2048
#define CCH  2048
#define DD   128
#define BT   (NB * TT)     // 8192 rows of x
#define N3C  (3 * CCH)     // 6144

#if defined(__AMDGCN__) && \
    __has_builtin(__builtin_amdgcn_global_load_async_to_lds_b128) && \
    __has_builtin(__builtin_amdgcn_s_wait_asynccnt)
#define HAVE_ASYNC_LDS 1
#else
#define HAVE_ASYNC_LDS 0
#endif

#if HAVE_ASYNC_LDS
// Builtin signature (probed via diagnostics): first arg is a pointer to
// int __attribute__((vector_size(16))) in addrspace(1); second in addrspace(3).
typedef int vint4 __attribute__((vector_size(16)));
typedef __attribute__((address_space(1))) vint4 as1_vint4;
typedef __attribute__((address_space(3))) vint4 as3_vint4;

__device__ __forceinline__ void async_copy_b128(const void* g, void* l) {
  // Cross-address-space casts are rejected by clang, so round-trip through
  // integers: global generic == as1 value; LDS offset == addr[31:0].
  as1_vint4* gp = (as1_vint4*)(unsigned long long)(size_t)g;
  as3_vint4* lp = (as3_vint4*)(unsigned int)(size_t)l;
  __builtin_amdgcn_global_load_async_to_lds_b128(gp, lp, 0, 0);
}
__device__ __forceinline__ void async_wait0() {
  __builtin_amdgcn_s_wait_asynccnt(0);
}
#endif

static __device__ __forceinline__ v8f wmma_f16(v16h a, v16h b, v8f c) {
  // 8 args: (neg_a, A, neg_b, B, c_mod, C, reuse_a, reuse_b)
  return __builtin_amdgcn_wmma_f32_16x16x32_f16(false, a, false, b, (short)0, c,
                                                false, false);
}

// ---------------------------------------------------------------------------
// Kernel 1: qkv = x @ w_attn + b_attn, scattered to f16 Q/K/V in [B,H,T,D].
// 128x128 output tile per block, BK=32. 8 waves arranged 4(M) x 2(N);
// each wave computes 32x64 via 2x4 WMMA 16x16 tiles.
// ---------------------------------------------------------------------------
__global__ __launch_bounds__(256)
void qkv_gemm_kernel(const float* __restrict__ x, const float* __restrict__ w,
                     const float* __restrict__ bias,
                     _Float16* __restrict__ qb, _Float16* __restrict__ kb,
                     _Float16* __restrict__ vb) {
  __shared__ __align__(16) _Float16 As [128 * 40];  // [row][k], stride 40
  __shared__ __align__(16) _Float16 BsT[128 * 40];  // [col][k], stride 40

  const int tid   = threadIdx.x;
  const int lane  = tid & 31;
  const int wv    = tid >> 5;
  const int khalf = lane >> 4;     // 0/1: which K half this lane holds
  const int n16   = lane & 15;
  const int wm    = wv >> 1;       // 0..3
  const int wn    = wv & 1;        // 0..1
  const int tM    = blockIdx.y * 128;
  const int tN    = blockIdx.x * 128;

  v8f acc[2][4];
#pragma unroll
  for (int mi = 0; mi < 2; ++mi)
#pragma unroll
    for (int ni = 0; ni < 4; ++ni)
#pragma unroll
      for (int r = 0; r < 8; ++r) acc[mi][ni][r] = 0.f;

  for (int k0 = 0; k0 < CCH; k0 += 32) {
    // hint-prefetch next k-tile while this one is staged/computed
    if (k0 + 32 < CCH) {
      __builtin_prefetch(x + (size_t)(tM + (tid >> 1)) * CCH + k0 + 32 + (tid & 1) * 16, 0, 0);
      __builtin_prefetch(w + (size_t)(k0 + 32 + (tid >> 3)) * N3C + tN + (tid & 7) * 16, 0, 0);
    }
    // X tile 128x32 fp32 -> f16 LDS (row major)
#pragma unroll
    for (int i = 0; i < 4; ++i) {
      int idx = tid + i * 256;            // 1024 float4 groups
      int r = idx >> 3, c4 = idx & 7;
      const float4 f = *(const float4*)(x + (size_t)(tM + r) * CCH + k0 + c4 * 4);
      _Float16* d = &As[r * 40 + c4 * 4];
      d[0] = (_Float16)f.x; d[1] = (_Float16)f.y;
      d[2] = (_Float16)f.z; d[3] = (_Float16)f.w;
    }
    // W tile 32x128 fp32 -> f16 LDS transposed so B-fragments are contiguous
#pragma unroll
    for (int i = 0; i < 4; ++i) {
      int idx = tid + i * 256;
      int r = idx >> 5, c4 = idx & 31;
      const float4 f = *(const float4*)(w + (size_t)(k0 + r) * N3C + tN + c4 * 4);
      BsT[(c4 * 4 + 0) * 40 + r] = (_Float16)f.x;
      BsT[(c4 * 4 + 1) * 40 + r] = (_Float16)f.y;
      BsT[(c4 * 4 + 2) * 40 + r] = (_Float16)f.z;
      BsT[(c4 * 4 + 3) * 40 + r] = (_Float16)f.w;
    }
    __syncthreads();

    v16h a[2];
#pragma unroll
    for (int mi = 0; mi < 2; ++mi) {
      // A layout: lane holds row M=n16; element j -> K = khalf*16 + j
      const _Float16* ap = &As[(wm * 32 + mi * 16 + n16) * 40 + khalf * 16];
#pragma unroll
      for (int j = 0; j < 16; ++j) a[mi][j] = ap[j];
    }
#pragma unroll
    for (int ni = 0; ni < 4; ++ni) {
      v16h b;
      // B layout: lane holds col N=n16; element j -> K = khalf*16 + j
      const _Float16* bp = &BsT[(wn * 64 + ni * 16 + n16) * 40 + khalf * 16];
#pragma unroll
      for (int j = 0; j < 16; ++j) b[j] = bp[j];
#pragma unroll
      for (int mi = 0; mi < 2; ++mi) acc[mi][ni] = wmma_f16(a[mi], b, acc[mi][ni]);
    }
    __syncthreads();
  }

  // Epilogue: +bias, route columns to Q/K/V with [B,H,T,D] layout (f16)
#pragma unroll
  for (int mi = 0; mi < 2; ++mi)
#pragma unroll
    for (int ni = 0; ni < 4; ++ni)
#pragma unroll
      for (int r = 0; r < 8; ++r) {
        int grow = tM + wm * 32 + mi * 16 + r + khalf * 8;   // C/D: M=r+8*(lane/16)
        int gcol = tN + wn * 64 + ni * 16 + n16;             // N=lane%16
        float val = acc[mi][ni][r] + bias[gcol];
        int which = gcol >> 11;          // 0:q 1:k 2:v
        int cc = gcol & 2047;
        int h = cc >> 7, d = cc & 127;
        int bb = grow >> 11, t = grow & 2047;
        _Float16* dst = (which == 0) ? qb : (which == 1) ? kb : vb;
        dst[((size_t)(bb * NH + h) * TT + t) * DD + d] = (_Float16)val;
      }
}

// ---------------------------------------------------------------------------
// Kernel 2: flash attention. Block = (128 q-rows, one (b,h)). 8 waves, each
// owns one 16-row M tile. Q fragments in registers; K and V^T in LDS; online
// softmax with width-16 shuffles; P staged through per-wave LDS to reformat
// accumulator (C layout) into A-fragment layout for the PV WMMA.
// K tile is a pure byte-copy -> async LDS DMA when available; V needs a
// transpose so it stays on the VGPR path (overlapping the async K copy).
// ---------------------------------------------------------------------------
__global__ __launch_bounds__(256)
void attn_kernel(const _Float16* __restrict__ qb, const _Float16* __restrict__ kb,
                 const _Float16* __restrict__ vb, _Float16* __restrict__ yb) {
  __shared__ __align__(16) _Float16 Ks [64 * 136];   // [key][d]
  __shared__ __align__(16) _Float16 VsT[128 * 72];   // [d][key]
  __shared__ __align__(16) _Float16 Ps [8][16 * 72]; // per-wave P scratch

  const int tid   = threadIdx.x;
  const int lane  = tid & 31;
  const int wv    = tid >> 5;
  const int khalf = lane >> 4;
  const int n16   = lane & 15;
  const int q0    = blockIdx.x * 128;
  const int bh    = blockIdx.y;
  const int bi    = bh >> 4, h = bh & 15;
  const size_t base = (size_t)bh * TT * DD;

  // Q fragments (A-matrix, 4 k-steps of 32 over D=128) held in registers
  const int qrow = q0 + wv * 16 + n16;
  v16h aq[4];
#pragma unroll
  for (int kk = 0; kk < 4; ++kk) {
    const _Float16* qp = qb + base + (size_t)qrow * DD + kk * 32 + khalf * 16;
#pragma unroll
    for (int j = 0; j < 16; ++j) aq[kk][j] = qp[j];
  }

  v8f oacc[8];
#pragma unroll
  for (int dt = 0; dt < 8; ++dt)
#pragma unroll
    for (int r = 0; r < 8; ++r) oacc[dt][r] = 0.f;
  float mrow[8], lrow[8];
#pragma unroll
  for (int r = 0; r < 8; ++r) { mrow[r] = -1e30f; lrow[r] = 0.f; }

  const float scale = 0.08838834764831845f;  // 1/sqrt(128)
  const int nkv = (q0 >> 6) + 2;             // causal: kv tiles covering q tile

  for (int kt = 0; kt < nkv; ++kt) {
    const int kv0 = kt * 64;
    __syncthreads();
    // K tile 64x128 (f16) -> LDS row major (pure copy)
#if HAVE_ASYNC_LDS
#pragma unroll
    for (int i = 0; i < 4; ++i) {
      int idx = tid + i * 256;                // 1024 b128 transfers
      int r = idx >> 4, c8 = idx & 15;
      async_copy_b128(kb + base + (size_t)(kv0 + r) * DD + c8 * 8,
                      &Ks[r * 136 + c8 * 8]);
    }
#else
#pragma unroll
    for (int i = 0; i < 4; ++i) {
      int idx = tid + i * 256;
      int r = idx >> 4, c8 = idx & 15;
      *(uint4*)&Ks[r * 136 + c8 * 8] =
          *(const uint4*)(kb + base + (size_t)(kv0 + r) * DD + c8 * 8);
    }
#endif
    // V tile 64x128 -> LDS transposed [d][key] (overlaps async K copy)
#pragma unroll
    for (int i = 0; i < 4; ++i) {
      int idx = tid + i * 256;
      int r = idx >> 4, c8 = idx & 15;
      uint4 u = *(const uint4*)(vb + base + (size_t)(kv0 + r) * DD + c8 * 8);
      const _Float16* e = (const _Float16*)&u;
#pragma unroll
      for (int j = 0; j < 8; ++j) VsT[(c8 * 8 + j) * 72 + r] = e[j];
    }
    // prefetch next kv tile (K and V) into cache while we compute
    if (kt + 1 < nkv) {
      __builtin_prefetch(kb + base + (size_t)(kv0 + 64) * DD + tid * 32, 0, 0);
      __builtin_prefetch(vb + base + (size_t)(kv0 + 64) * DD + tid * 32, 0, 0);
    }
#if HAVE_ASYNC_LDS
    async_wait0();
#endif
    __syncthreads();

    // S = Q K^T  (16 x 64 per wave)
    v8f s[4];
#pragma unroll
    for (int ni = 0; ni < 4; ++ni)
#pragma unroll
      for (int r = 0; r < 8; ++r) s[ni][r] = 0.f;
#pragma unroll
    for (int kk = 0; kk < 4; ++kk) {
#pragma unroll
      for (int ni = 0; ni < 4; ++ni) {
        v16h bk;  // B element j = K[key = ni*16+n16][d = kk*32 + khalf*16 + j]
        const _Float16* bp = &Ks[(ni * 16 + n16) * 136 + kk * 32 + khalf * 16];
#pragma unroll
        for (int j = 0; j < 16; ++j) bk[j] = bp[j];
        s[ni] = wmma_f16(aq[kk], bk, s[ni]);
      }
    }

    // scale + causal mask
    const bool need_mask = (kv0 + 63) > (q0 + wv * 16);
#pragma unroll
    for (int ni = 0; ni < 4; ++ni) {
      int col = kv0 + ni * 16 + n16;
#pragma unroll
      for (int r = 0; r < 8; ++r) {
        int row = q0 + wv * 16 + r + khalf * 8;
        float sv = s[ni][r] * scale;
        if (need_mask && col > row) sv = -1e30f;
        s[ni][r] = sv;
      }
    }

    // online softmax (row = (r, khalf); reduce across 16-lane groups)
#pragma unroll
    for (int r = 0; r < 8; ++r) {
      float mx = s[0][r];
#pragma unroll
      for (int ni = 1; ni < 4; ++ni) mx = fmaxf(mx, s[ni][r]);
#pragma unroll
      for (int off = 1; off < 16; off <<= 1) mx = fmaxf(mx, __shfl_xor(mx, off, 16));
      float mn   = fmaxf(mrow[r], mx);
      float corr = __expf(mrow[r] - mn);
      mrow[r] = mn;
      lrow[r] *= corr;
#pragma unroll
      for (int dt = 0; dt < 8; ++dt) oacc[dt][r] *= corr;
      float ps = 0.f;
#pragma unroll
      for (int ni = 0; ni < 4; ++ni) {
        float p = __expf(s[ni][r] - mn);
        ps += p;
        s[ni][r] = p;
      }
#pragma unroll
      for (int off = 1; off < 16; off <<= 1) ps += __shfl_xor(ps, off, 16);
      lrow[r] += ps;
    }

    // stash P (C layout) into per-wave LDS so it can be re-read in A layout
    _Float16* pw = &Ps[wv][0];
#pragma unroll
    for (int ni = 0; ni < 4; ++ni)
#pragma unroll
      for (int r = 0; r < 8; ++r)
        pw[(r + khalf * 8) * 72 + ni * 16 + n16] = (_Float16)s[ni][r];

    // O += P @ V   (K-dim = 64 keys -> 2 WMMA k-steps)
#pragma unroll
    for (int ks = 0; ks < 2; ++ks) {
      v16h ap;
      const _Float16* app = &pw[n16 * 72 + ks * 32 + khalf * 16];
#pragma unroll
      for (int j = 0; j < 16; ++j) ap[j] = app[j];
#pragma unroll
      for (int dt = 0; dt < 8; ++dt) {
        v16h bv;  // B element j = V[key = ks*32 + khalf*16 + j][d = dt*16+n16]
        const _Float16* bvp = &VsT[(dt * 16 + n16) * 72 + ks * 32 + khalf * 16];
#pragma unroll
        for (int j = 0; j < 16; ++j) bv[j] = bvp[j];
        oacc[dt] = wmma_f16(ap, bv, oacc[dt]);
      }
    }
  }

  // finalize: O /= l ; write y as f16 [B,T,C] with c = h*128 + d
#pragma unroll
  for (int r = 0; r < 8; ++r) {
    float inv = 1.f / lrow[r];
    int t = q0 + wv * 16 + r + khalf * 8;
#pragma unroll
    for (int dt = 0; dt < 8; ++dt) {
      int c = h * DD + dt * 16 + n16;
      yb[((size_t)bi * TT + t) * CCH + c] = (_Float16)(oacc[dt][r] * inv);
    }
  }
}

// ---------------------------------------------------------------------------
// Kernel 3: out = y @ w_proj + b_proj (fp32 output). Same scheme as kernel 1
// but A is already f16 -> pure copy, eligible for async LDS DMA.
// ---------------------------------------------------------------------------
__global__ __launch_bounds__(256)
void proj_gemm_kernel(const _Float16* __restrict__ yb, const float* __restrict__ w,
                      const float* __restrict__ bias, float* __restrict__ out) {
  __shared__ __align__(16) _Float16 As [128 * 40];
  __shared__ __align__(16) _Float16 BsT[128 * 40];

  const int tid   = threadIdx.x;
  const int lane  = tid & 31;
  const int wv    = tid >> 5;
  const int khalf = lane >> 4;
  const int n16   = lane & 15;
  const int wm    = wv >> 1;
  const int wn    = wv & 1;
  const int tM    = blockIdx.y * 128;
  const int tN    = blockIdx.x * 128;

  v8f acc[2][4];
#pragma unroll
  for (int mi = 0; mi < 2; ++mi)
#pragma unroll
    for (int ni = 0; ni < 4; ++ni)
#pragma unroll
      for (int r = 0; r < 8; ++r) acc[mi][ni][r] = 0.f;

  for (int k0 = 0; k0 < CCH; k0 += 32) {
    if (k0 + 32 < CCH) {
      __builtin_prefetch(yb + (size_t)(tM + (tid >> 1)) * CCH + k0 + 32 + (tid & 1) * 16, 0, 0);
      __builtin_prefetch(w + (size_t)(k0 + 32 + (tid >> 3)) * CCH + tN + (tid & 7) * 16, 0, 0);
    }
    // A tile 128x32 f16: pure copy -> async LDS DMA when available
#if HAVE_ASYNC_LDS
#pragma unroll
    for (int i = 0; i < 2; ++i) {
      int idx = tid + i * 256;                 // 512 b128 transfers
      int r = idx >> 2, c8 = idx & 3;
      async_copy_b128(yb + (size_t)(tM + r) * CCH + k0 + c8 * 8,
                      &As[r * 40 + c8 * 8]);
    }
#else
#pragma unroll
    for (int i = 0; i < 2; ++i) {
      int idx = tid + i * 256;
      int r = idx >> 2, c8 = idx & 3;
      *(uint4*)&As[r * 40 + c8 * 8] =
          *(const uint4*)(yb + (size_t)(tM + r) * CCH + k0 + c8 * 8);
    }
#endif
    // B tile 32x128 fp32 -> f16 transposed (overlaps async A copy)
#pragma unroll
    for (int i = 0; i < 4; ++i) {
      int idx = tid + i * 256;
      int r = idx >> 5, c4 = idx & 31;
      const float4 f = *(const float4*)(w + (size_t)(k0 + r) * CCH + tN + c4 * 4);
      BsT[(c4 * 4 + 0) * 40 + r] = (_Float16)f.x;
      BsT[(c4 * 4 + 1) * 40 + r] = (_Float16)f.y;
      BsT[(c4 * 4 + 2) * 40 + r] = (_Float16)f.z;
      BsT[(c4 * 4 + 3) * 40 + r] = (_Float16)f.w;
    }
#if HAVE_ASYNC_LDS
    async_wait0();
#endif
    __syncthreads();

    v16h a[2];
#pragma unroll
    for (int mi = 0; mi < 2; ++mi) {
      const _Float16* ap = &As[(wm * 32 + mi * 16 + n16) * 40 + khalf * 16];
#pragma unroll
      for (int j = 0; j < 16; ++j) a[mi][j] = ap[j];
    }
#pragma unroll
    for (int ni = 0; ni < 4; ++ni) {
      v16h b;
      const _Float16* bp = &BsT[(wn * 64 + ni * 16 + n16) * 40 + khalf * 16];
#pragma unroll
      for (int j = 0; j < 16; ++j) b[j] = bp[j];
#pragma unroll
      for (int mi = 0; mi < 2; ++mi) acc[mi][ni] = wmma_f16(a[mi], b, acc[mi][ni]);
    }
    __syncthreads();
  }

#pragma unroll
  for (int mi = 0; mi < 2; ++mi)
#pragma unroll
    for (int ni = 0; ni < 4; ++ni)
#pragma unroll
      for (int r = 0; r < 8; ++r) {
        int grow = tM + wm * 32 + mi * 16 + r + khalf * 8;
        int gcol = tN + wn * 64 + ni * 16 + n16;
        out[(size_t)grow * CCH + gcol] = acc[mi][ni][r] + bias[gcol];
      }
}

// ---------------------------------------------------------------------------
extern "C" void kernel_launch(void* const* d_in, const int* in_sizes, int n_in,
                              void* d_out, int out_size, void* d_ws, size_t ws_size,
                              hipStream_t stream) {
  const float* x      = (const float*)d_in[0];
  const float* w_attn = (const float*)d_in[1];
  const float* b_attn = (const float*)d_in[2];
  const float* w_proj = (const float*)d_in[3];
  const float* b_proj = (const float*)d_in[4];
  float* out = (float*)d_out;

  const size_t QSZ = (size_t)NB * NH * TT * DD;  // elements per Q/K/V buffer
  _Float16* qb = (_Float16*)d_ws;
  _Float16* kb = qb + QSZ;
  _Float16* vb = kb + QSZ;
  _Float16* yb = vb + QSZ;  // [B*T, C] f16 attention output

  qkv_gemm_kernel<<<dim3(N3C / 128, BT / 128), 256, 0, stream>>>(
      x, w_attn, b_attn, qb, kb, vb);
  attn_kernel<<<dim3(TT / 128, NB * NH), 256, 0, stream>>>(qb, kb, vb, yb);
  proj_gemm_kernel<<<dim3(CCH / 128, BT / 128), 256, 0, stream>>>(
      yb, w_proj, b_proj, out);
}
